// Rwkv_44375602102284
// MI455X (gfx1250) — matrix operand order repaired and checked by reading, not compiled, not run on hardware
//
#include <hip/hip_runtime.h>
#include <hip/hip_bf16.h>
#include <stdint.h>

typedef __attribute__((ext_vector_type(16))) __bf16 v16bf;
typedef __attribute__((ext_vector_type(8)))  __bf16 v8bf;
typedef __attribute__((ext_vector_type(8)))  float  v8f;

#define Bn 16
#define Tn 2048
#define Cn 768

static __device__ __forceinline__ unsigned short f2bf(float f) {
    union { float f; uint32_t u; } x; x.f = f;
    uint32_t r = x.u + 0x7FFFu + ((x.u >> 16) & 1u);   // round-to-nearest-even
    return (unsigned short)(r >> 16);
}
static __device__ __forceinline__ float bf2f(unsigned short h) {
    union { uint32_t u; float f; } x; x.u = ((uint32_t)h) << 16;
    return x.f;
}

// ---- 1) cast the four projection weights fp32 -> bf16 (L2-resident afterwards) ----
__global__ void wconv_kernel(const float* __restrict__ w0, const float* __restrict__ w1,
                             const float* __restrict__ w2, const float* __restrict__ w3,
                             unsigned short* __restrict__ dst) {
    const int which = blockIdx.y;
    const float* s = (which == 0) ? w0 : (which == 1) ? w1 : (which == 2) ? w2 : w3;
    int i = blockIdx.x * 256 + threadIdx.x;
    if (i < Cn * Cn) dst[(size_t)which * (Cn * Cn) + i] = f2bf(s[i]);
}

// ---- 2) token shift + mix, emit xk/xv/xr in bf16 ----
__global__ void mix_kernel(const float* __restrict__ x,
                           const float* __restrict__ mk, const float* __restrict__ mv,
                           const float* __restrict__ mr,
                           unsigned short* __restrict__ xk, unsigned short* __restrict__ xv,
                           unsigned short* __restrict__ xr) {
    long long i = (long long)blockIdx.x * 256 + threadIdx.x;
    if (i >= (long long)Bn * Tn * Cn) return;
    int c = (int)(i % Cn);
    long long tc = i % ((long long)Tn * Cn);
    float xc = x[i];
    float xp = (tc < Cn) ? 0.0f : x[i - Cn];   // x_prev, zero at t==0 per batch
    float a;
    a = mk[c]; xk[i] = f2bf(xc * a + xp * (1.0f - a));
    a = mv[c]; xv[i] = f2bf(xc * a + xp * (1.0f - a));
    a = mr[c]; xr[i] = f2bf(xc * a + xp * (1.0f - a));
}

// ---- 3) bf16 WMMA GEMM: out[m,n] = sum_k A[m,k]*W[n,k] ----
// Wave tile 64x64 = 4x4 WMMA 16x16x32 tiles; 8 waves/block -> block tile 512(M)x64(N).
// MODE 0: fp32 store; MODE 1: sigmoid -> bf16 store (for the r projection).
template<int MODE>
__global__ void __launch_bounds__(256) gemm_bf16_wmma(
    const unsigned short* __restrict__ A,   // [M,K] bf16 row-major, streamed
    const unsigned short* __restrict__ W,   // [N,K] bf16 row-major (B = W^T), L2-resident
    float* __restrict__ outF,
    unsigned short* __restrict__ outH)
{
    constexpr int N = Cn, K = Cn;
    const int lane = threadIdx.x & 31;
    const int wave = threadIdx.x >> 5;
    const int hlf  = lane >> 4;      // half-wave select (K-group: k = 16*hlf + 0..15)
    const int lrow = lane & 15;      // row (A: M) / col (B: N) within tile
    const int nBase = blockIdx.x * 64;
    const int mBase = blockIdx.y * 512 + wave * 64;

    v8f acc[4][4] = {};

#pragma unroll 2
    for (int kk = 0; kk < K; kk += 32) {
        // B fragment (32x16 bf16): half 0 holds K 0..15 contiguous, half 1 K 16..31
        v16bf bfrag[4];
#pragma unroll
        for (int j = 0; j < 4; ++j) {
            const unsigned short* wp = W + (size_t)(nBase + j * 16 + lrow) * K + kk + hlf * 16;
            v8bf b0 = *(const v8bf*)(wp);
            v8bf b1 = *(const v8bf*)(wp + 8);
            bfrag[j] = __builtin_shufflevector(b0, b1, 0,1,2,3,4,5,6,7,8,9,10,11,12,13,14,15);
        }
        // A fragment (16x32 bf16): half 0 holds K 0..15 contiguous, half 1 K 16..31
        v16bf afrag[4];
#pragma unroll
        for (int i = 0; i < 4; ++i) {
            const unsigned short* ap = A + (size_t)(mBase + i * 16 + lrow) * K + kk + hlf * 16;
            v8bf a0 = *(const v8bf*)(ap);
            v8bf a1 = *(const v8bf*)(ap + 8);
            afrag[i] = __builtin_shufflevector(a0, a1, 0,1,2,3,4,5,6,7,8,9,10,11,12,13,14,15);
            // Unconditional speculative prefetch one cacheline (2 K-steps) ahead:
            // branch-free; OOB prefetch is dropped silently, no LOADcnt impact.
            __builtin_prefetch(ap + 64, 0, 3);
        }
#pragma unroll
        for (int i = 0; i < 4; ++i)
#pragma unroll
            for (int j = 0; j < 4; ++j)
                acc[i][j] = __builtin_amdgcn_wmma_f32_16x16x32_bf16(
                    false, afrag[i], false, bfrag[j], (short)0, acc[i][j], false, false);
    }

    // C/D layout: lane(0..15)=N col, half selects M+8, VGPR r = M row within half
#pragma unroll
    for (int i = 0; i < 4; ++i) {
#pragma unroll
        for (int j = 0; j < 4; ++j) {
            const int n  = nBase + j * 16 + lrow;
            const int m0 = mBase + i * 16 + hlf * 8;
#pragma unroll
            for (int r = 0; r < 8; ++r) {
                float val = acc[i][j][r];
                size_t o = (size_t)(m0 + r) * N + n;
                if (MODE == 1) outH[o] = f2bf(1.0f / (1.0f + __expf(-val)));
                else           outF[o] = val;
            }
        }
    }
}

// ---- 4) numerically-stable WKV recurrence, fused with r-gate, bf16 out ----
__global__ void wkv_scan(const float* __restrict__ k, const float* __restrict__ v,
                         const unsigned short* __restrict__ r,
                         const float* __restrict__ time_decay,
                         const float* __restrict__ time_first,
                         unsigned short* __restrict__ rwkv)
{
    int gid = blockIdx.x * 256 + threadIdx.x;
    if (gid >= Bn * Cn) return;
    const int b = gid / Cn, c = gid % Cn;
    const float w = -__expf(time_decay[c]);
    const float u = time_first[c];
    float num = 0.0f, den = 0.0f, mx = -1e38f;
    const size_t base = (size_t)b * Tn * Cn + c;
    for (int t = 0; t < Tn; ++t) {
        const size_t idx = base + (size_t)t * Cn;   // consecutive lanes -> consecutive c
        const float kt = k[idx], vt = v[idx];
        const float ku = kt + u;
        const float mo = fmaxf(mx, ku);
        const float e1 = __expf(mx - mo);
        const float e2 = __expf(ku - mo);
        const float out = (e1 * num + e2 * vt) / (e1 * den + e2);
        rwkv[idx] = f2bf(bf2f(r[idx]) * out);
        const float mw = mx + w;
        const float mn = fmaxf(mw, kt);
        const float e1s = __expf(mw - mn);
        const float e2s = __expf(kt - mn);
        num = e1s * num + e2s * vt;
        den = e1s * den + e2s;
        mx = mn;
    }
}

extern "C" void kernel_launch(void* const* d_in, const int* in_sizes, int n_in,
                              void* d_out, int out_size, void* d_ws, size_t ws_size,
                              hipStream_t stream) {
    const float* x  = (const float*)d_in[0];
    const float* td = (const float*)d_in[1];
    const float* tf = (const float*)d_in[2];
    const float* mk = (const float*)d_in[3];
    const float* mv = (const float*)d_in[4];
    const float* mr = (const float*)d_in[5];
    const float* Wk = (const float*)d_in[6];
    const float* Wv = (const float*)d_in[7];
    const float* Wr = (const float*)d_in[8];
    const float* Wo = (const float*)d_in[9];
    float* out = (float*)d_out;

    const size_t CC  = (size_t)Cn * Cn;         // 589824
    const size_t BTC = (size_t)Bn * Tn * Cn;    // 25165824

    char* ws = (char*)d_ws;
    unsigned short* wb  = (unsigned short*)ws;             // 4*CC bf16 weights
    unsigned short* xk  = (unsigned short*)(ws + 4 * CC * 2);
    unsigned short* xv  = xk + BTC;
    unsigned short* xr  = xv + BTC;
    float* kbuf = (float*)(xr + BTC);
    float* vbuf = kbuf + BTC;
    unsigned short* rbuf = (unsigned short*)(vbuf + BTC);
    unsigned short* rwkv = xk;                  // xk dead after first GEMM -> reuse

    wconv_kernel<<<dim3((unsigned)((CC + 255) / 256), 4), 256, 0, stream>>>(Wk, Wv, Wr, Wo, wb);
    mix_kernel<<<(unsigned)((BTC + 255) / 256), 256, 0, stream>>>(x, mk, mv, mr, xk, xv, xr);

    dim3 g(Cn / 64, (Bn * Tn) / 512);           // (12, 64)
    gemm_bf16_wmma<0><<<g, 256, 0, stream>>>(xk, wb + 0 * CC, kbuf, nullptr);
    gemm_bf16_wmma<0><<<g, 256, 0, stream>>>(xv, wb + 1 * CC, vbuf, nullptr);
    gemm_bf16_wmma<1><<<g, 256, 0, stream>>>(xr, wb + 2 * CC, nullptr, rbuf);

    wkv_scan<<<(Bn * Cn + 255) / 256, 256, 0, stream>>>(kbuf, vbuf, rbuf, td, tf, rwkv);

    gemm_bf16_wmma<0><<<g, 256, 0, stream>>>(rwkv, wb + 3 * CC, out, nullptr);
}
